// PointNet2Backbone_24524263260777
// MI455X (gfx1250) — matrix operand order repaired and checked by reading, not compiled, not run on hardware
//
#include <hip/hip_runtime.h>

typedef __attribute__((ext_vector_type(16))) _Float16 v16h;
typedef __attribute__((ext_vector_type(8)))  float    v8f;

#define B_SZ 2
#define N0   16384
#define KNB  32

// ---------------------------------------------------------------------------
// Split points (B,N,4) -> xyz (B,N,3), feat (B,N,1)
// ---------------------------------------------------------------------------
__global__ void split_points_kernel(const float* __restrict__ pts, int N,
                                    float* __restrict__ xyz, float* __restrict__ feat) {
  int n = blockIdx.x * blockDim.x + threadIdx.x;
  int b = blockIdx.y;
  if (n >= N) return;
  const float* p = pts + ((size_t)b * N + n) * 4;
  float* x = xyz + ((size_t)b * N + n) * 3;
  x[0] = p[0]; x[1] = p[1]; x[2] = p[2];
  feat[(size_t)b * N + n] = p[3];
}

// ---------------------------------------------------------------------------
// Farthest point sampling: one block per batch, distances in registers.
// ---------------------------------------------------------------------------
#define FPS_T 256
__global__ void fps_kernel(const float* __restrict__ xyz, int N, int npoint,
                           int* __restrict__ out_idx) {
  int b = blockIdx.x;
  const float* p = xyz + (size_t)b * N * 3;
  __shared__ float s_val[FPS_T];
  __shared__ int   s_idx[FPS_T];
  __shared__ int   s_far;
  float dist[64];
  int per = (N + FPS_T - 1) / FPS_T;
  if (per > 64) per = 64;
  for (int t = 0; t < per; ++t) dist[t] = 1e10f;
  if (threadIdx.x == 0) s_far = 0;
  __syncthreads();
  for (int it = 0; it < npoint; ++it) {
    int far = s_far;
    if (threadIdx.x == 0) out_idx[b * npoint + it] = far;
    float cx = p[far * 3 + 0], cy = p[far * 3 + 1], cz = p[far * 3 + 2];
    float best = -1.f; int besti = 0;
    for (int t = 0; t < per; ++t) {
      int i = threadIdx.x + t * FPS_T;
      if (i < N) {
        float dx = p[i * 3 + 0] - cx;
        float dy = p[i * 3 + 1] - cy;
        float dz = p[i * 3 + 2] - cz;
        float d = dx * dx + dy * dy + dz * dz;
        if (d < dist[t]) dist[t] = d;
        if (dist[t] > best) { best = dist[t]; besti = i; }
      }
    }
    s_val[threadIdx.x] = best; s_idx[threadIdx.x] = besti;
    __syncthreads();
    for (int s = FPS_T / 2; s > 0; s >>= 1) {
      if (threadIdx.x < s) {
        if (s_val[threadIdx.x + s] > s_val[threadIdx.x]) {
          s_val[threadIdx.x] = s_val[threadIdx.x + s];
          s_idx[threadIdx.x] = s_idx[threadIdx.x + s];
        }
      }
      __syncthreads();
    }
    if (threadIdx.x == 0) s_far = s_idx[0];
    __syncthreads();
  }
}

// ---------------------------------------------------------------------------
// Gather sampled centroids
// ---------------------------------------------------------------------------
__global__ void gather_xyz_kernel(const float* __restrict__ xyz, const int* __restrict__ idx,
                                  int N, int S, float* __restrict__ new_xyz) {
  int s = blockIdx.x * blockDim.x + threadIdx.x;
  int b = blockIdx.y;
  if (s >= S) return;
  int id = idx[b * S + s];
  for (int c = 0; c < 3; ++c)
    new_xyz[((size_t)b * S + s) * 3 + c] = xyz[((size_t)b * N + id) * 3 + c];
}

// ---------------------------------------------------------------------------
// Ball query: first K in-radius indices in ascending index order, pad w/ first
// ---------------------------------------------------------------------------
__global__ void ballquery_kernel(const float* __restrict__ xyz, const float* __restrict__ nxyz,
                                 int N, int S, int K, float r2, int* __restrict__ idx_out) {
  int s = blockIdx.x * blockDim.x + threadIdx.x;
  int b = blockIdx.y;
  if (s >= S) return;
  const float* p = xyz + (size_t)b * N * 3;
  float cx = nxyz[((size_t)b * S + s) * 3 + 0];
  float cy = nxyz[((size_t)b * S + s) * 3 + 1];
  float cz = nxyz[((size_t)b * S + s) * 3 + 2];
  int* out = idx_out + ((size_t)b * S + s) * K;
  int cnt = 0, first = 0; bool have = false;
  for (int i = 0; i < N && cnt < K; ++i) {
    float dx = p[i * 3 + 0] - cx;
    float dy = p[i * 3 + 1] - cy;
    float dz = p[i * 3 + 2] - cz;
    float d = dx * dx + dy * dy + dz * dz;
    if (d <= r2) {
      if (!have) { first = i; have = true; }
      out[cnt++] = i;
    }
  }
  for (; cnt < K; ++cnt) out[cnt] = first;
}

// ---------------------------------------------------------------------------
// Build grouped input X (B, 3+Cf, S*K): recentered xyz then gathered features
// ---------------------------------------------------------------------------
__global__ void group_kernel(const float* __restrict__ xyz, const float* __restrict__ feat,
                             const float* __restrict__ nxyz, const int* __restrict__ idx,
                             int N, int S, int K, int Cf, float* __restrict__ X) {
  int l = blockIdx.x * blockDim.x + threadIdx.x;  // s*K + k
  int b = blockIdx.y;
  int L = S * K;
  if (l >= L) return;
  int s = l / K;
  int id = idx[((size_t)b * S + s) * K + (l % K)];
  int Cin = 3 + Cf;
  float* xb = X + (size_t)b * Cin * L;
  for (int c = 0; c < 3; ++c)
    xb[(size_t)c * L + l] = xyz[((size_t)b * N + id) * 3 + c] - nxyz[((size_t)b * S + s) * 3 + c];
  for (int c = 0; c < Cf; ++c)
    xb[(size_t)(3 + c) * L + l] = feat[((size_t)b * N + id) * Cf + c];
}

// ---------------------------------------------------------------------------
// Pack W (f32, Cout x Cin) into f16 A-fragments, ISA lane layout, K zero-padded.
// Fragment (mt, ks): 32 lanes x 16 halfs contiguous.
// ---------------------------------------------------------------------------
__global__ void pack_w_kernel(const float* __restrict__ W, _Float16* __restrict__ Ap,
                              int Cin, int Cout, int nks) {
  int t = blockIdx.x * blockDim.x + threadIdx.x;  // (mt*nks + ks)*32 + lane
  int total = (Cout / 16) * nks * 32;
  if (t >= total) return;
  int lane = t & 31;
  int ks = (t >> 5) % nks;
  int mt = (t >> 5) / nks;
  int m = mt * 16 + (lane & 15);
  int hi = lane >> 4;
  _Float16* dst = Ap + (size_t)t * 16;
#pragma unroll
  for (int i = 0; i < 16; ++i) {
    int j = i >> 1, p = i & 1;
    int kk = (j < 4) ? (hi * 8 + 2 * j + p) : (16 + hi * 8 + 2 * (j - 4) + p);
    int k = ks * 32 + kk;
    dst[i] = (_Float16)((k < Cin) ? W[(size_t)m * Cin + k] : 0.f);
  }
}

// ---------------------------------------------------------------------------
// Pack X (f32, B x Cin x L) into f16 B-fragments, K zero-padded.
// Fragment (b, nt, ks): 32 lanes x 16 halfs contiguous; lane = khalf*16 + n.
// ---------------------------------------------------------------------------
__global__ void pack_x_kernel(const float* __restrict__ X, _Float16* __restrict__ Bp,
                              int Cin, int L, int nks) {
  int t = blockIdx.x * blockDim.x + threadIdx.x;  // (nt*nks + ks)*32 + lane
  int b = blockIdx.y;
  int total = (L / 16) * nks * 32;
  if (t >= total) return;
  int lane = t & 31;
  int ks = (t >> 5) % nks;
  int nt = (t >> 5) / nks;
  int n = nt * 16 + (lane & 15);
  int hi = lane >> 4;
  const float* Xb = X + (size_t)b * Cin * L;
  _Float16* dst = Bp + ((size_t)b * total + t) * 16;
#pragma unroll
  for (int i = 0; i < 16; ++i) {
    int k = ks * 32 + ((i < 8) ? (hi * 8 + i) : (16 + hi * 8 + (i - 8)));
    dst[i] = (_Float16)((k < Cin) ? Xb[(size_t)k * L + n] : 0.f);
  }
}

// ---------------------------------------------------------------------------
// WMMA GEMM on packed fragments: Y[b] (Cout x L) = W * X + bias.
// Each wave: 32x32 output tile = 2x2 register tiling, 4 WMMAs per K-slab.
// Branch-free inner loop: 4 contiguous 32B vector loads + 4 v_wmma.
// ---------------------------------------------------------------------------
__global__ void gemm_wmma_kernel(const _Float16* __restrict__ Ap,
                                 const _Float16* __restrict__ Bp,
                                 const float* __restrict__ bias,
                                 float* __restrict__ Y,
                                 int nks, int Cout, int L) {
  int lane = threadIdx.x & 31;
  int wave = threadIdx.x >> 5;
  int nt = (blockIdx.x * (blockDim.x >> 5) + wave) * 2;  // two 16-wide N tiles
  int mt = blockIdx.y * 2;                               // two 16-wide M tiles
  int b  = blockIdx.z;
  if (nt * 16 >= L) return;  // wave-uniform: EXEC all-ones for WMMA

  size_t bOff = (size_t)b * (size_t)(L / 16) * nks * 512;
  const _Float16* a0p = Ap + ((size_t)(mt + 0) * nks * 32 + lane) * 16;
  const _Float16* a1p = Ap + ((size_t)(mt + 1) * nks * 32 + lane) * 16;
  const _Float16* b0p = Bp + bOff + ((size_t)(nt + 0) * nks * 32 + lane) * 16;
  const _Float16* b1p = Bp + bOff + ((size_t)(nt + 1) * nks * 32 + lane) * 16;

  v8f acc00 = {}, acc01 = {}, acc10 = {}, acc11 = {};
  for (int ks = 0; ks < nks; ++ks) {
    v16h a0 = *(const v16h*)a0p;
    v16h a1 = *(const v16h*)a1p;
    v16h f0 = *(const v16h*)b0p;
    v16h f1 = *(const v16h*)b1p;
    a0p += 512; a1p += 512; b0p += 512; b1p += 512;
    if (ks + 1 < nks) {
      __builtin_prefetch(b0p, 0, 1);  // global_prefetch_b8
      __builtin_prefetch(b1p, 0, 1);
    }
    acc00 = __builtin_amdgcn_wmma_f32_16x16x32_f16(false, a0, false, f0, (short)0, acc00, false, false);
    acc01 = __builtin_amdgcn_wmma_f32_16x16x32_f16(false, a0, false, f1, (short)0, acc01, false, false);
    acc10 = __builtin_amdgcn_wmma_f32_16x16x32_f16(false, a1, false, f0, (short)0, acc10, false, false);
    acc11 = __builtin_amdgcn_wmma_f32_16x16x32_f16(false, a1, false, f1, (short)0, acc11, false, false);
  }

  int hi = lane >> 4, n = lane & 15;
  float* Yb = Y + (size_t)b * Cout * L;
  int n0 = nt * 16 + n;
#pragma unroll
  for (int r = 0; r < 8; ++r) {
    int m0 = mt * 16 + r + 8 * hi;
    int m1 = m0 + 16;
    float bv0 = bias[m0], bv1 = bias[m1];
    Yb[(size_t)m0 * L + n0]      = acc00[r] + bv0;
    Yb[(size_t)m0 * L + n0 + 16] = acc01[r] + bv0;
    Yb[(size_t)m1 * L + n0]      = acc10[r] + bv1;
    Yb[(size_t)m1 * L + n0 + 16] = acc11[r] + bv1;
  }
}

// ---------------------------------------------------------------------------
// BatchNorm statistics: per-channel mean/var over (B, L)
// ---------------------------------------------------------------------------
__global__ void bn_stats_kernel(const float* __restrict__ Y, float* __restrict__ mean,
                                float* __restrict__ var, int Cout, int L, int B) {
  int c = blockIdx.x;
  __shared__ float s_s[256], s_q[256];
  float s = 0.f, q = 0.f;
  for (int b = 0; b < B; ++b) {
    const float* y = Y + ((size_t)b * Cout + c) * L;
    for (int i = threadIdx.x; i < L; i += blockDim.x) {
      float v = y[i]; s += v; q += v * v;
    }
  }
  s_s[threadIdx.x] = s; s_q[threadIdx.x] = q;
  __syncthreads();
  for (int t = 128; t > 0; t >>= 1) {
    if (threadIdx.x < t) {
      s_s[threadIdx.x] += s_s[threadIdx.x + t];
      s_q[threadIdx.x] += s_q[threadIdx.x + t];
    }
    __syncthreads();
  }
  if (threadIdx.x == 0) {
    float inv = 1.f / (float)(B * L);
    float m = s_s[0] * inv;
    mean[c] = m;
    var[c] = s_q[0] * inv - m * m;
  }
}

__global__ void bn_relu_kernel(float* __restrict__ Y, const float* __restrict__ mean,
                               const float* __restrict__ var, const float* __restrict__ gamma,
                               const float* __restrict__ beta, int Cout, int L,
                               unsigned long long total) {
  unsigned long long i = (unsigned long long)blockIdx.x * blockDim.x + threadIdx.x;
  if (i >= total) return;
  int c = (int)((i / (unsigned long long)L) % (unsigned long long)Cout);
  float x = Y[i];
  x = (x - mean[c]) * rsqrtf(var[c] + 1e-5f);
  x = x * gamma[c] + beta[c];
  Y[i] = fmaxf(x, 0.f);
}

// ---------------------------------------------------------------------------
// Max over K neighbors: Y (B,C,S*K) -> feat (B,S,C)
// ---------------------------------------------------------------------------
__global__ void maxpool_kernel(const float* __restrict__ Y, float* __restrict__ feat,
                               int C, int S, int K) {
  int t = blockIdx.x * blockDim.x + threadIdx.x;
  int b = blockIdx.y;
  if (t >= S * C) return;
  int s = t / C, c = t % C;
  const float* y = Y + ((size_t)b * C + c) * (size_t)(S * K) + (size_t)s * K;
  float m = y[0];
  for (int k = 1; k < K; ++k) m = fmaxf(m, y[k]);
  feat[((size_t)b * S + s) * C + c] = m;
}

// Global SA concat: X (B, 3+Cf, N)
__global__ void concat_xyzfeat_kernel(const float* __restrict__ xyz, const float* __restrict__ feat,
                                      int N, int Cf, float* __restrict__ X) {
  int n = blockIdx.x * blockDim.x + threadIdx.x;
  int b = blockIdx.y;
  if (n >= N) return;
  int Cin = 3 + Cf;
  float* xb = X + (size_t)b * Cin * N;
  for (int c = 0; c < 3; ++c) xb[(size_t)c * N + n] = xyz[((size_t)b * N + n) * 3 + c];
  for (int c = 0; c < Cf; ++c) xb[(size_t)(3 + c) * N + n] = feat[((size_t)b * N + n) * Cf + c];
}

// Max over all L positions: Y (B,C,L) -> out (B,C)
__global__ void chanmax_kernel(const float* __restrict__ Y, float* __restrict__ out,
                               int C, int L) {
  int c = blockIdx.x * blockDim.x + threadIdx.x;
  int b = blockIdx.y;
  if (c >= C) return;
  const float* y = Y + ((size_t)b * C + c) * L;
  float m = y[0];
  for (int i = 1; i < L; ++i) m = fmaxf(m, y[i]);
  out[(size_t)b * C + c] = m;
}

// ---------------------------------------------------------------------------
// 3-NN inverse-distance interpolation (S==1 -> broadcast)
// ---------------------------------------------------------------------------
__global__ void interp_kernel(const float* __restrict__ xyz1, const float* __restrict__ xyz2,
                              const float* __restrict__ feat2, int N, int S, int C2,
                              float* __restrict__ interp) {
  int n = blockIdx.x * blockDim.x + threadIdx.x;
  int b = blockIdx.y;
  if (n >= N) return;
  const float* f2 = feat2 + (size_t)b * S * C2;
  float* out = interp + ((size_t)b * N + n) * C2;
  if (S == 1) {
    for (int c = 0; c < C2; ++c) out[c] = f2[c];
    return;
  }
  float px = xyz1[((size_t)b * N + n) * 3 + 0];
  float py = xyz1[((size_t)b * N + n) * 3 + 1];
  float pz = xyz1[((size_t)b * N + n) * 3 + 2];
  float d0 = 1e30f, d1 = 1e30f, d2 = 1e30f;
  int i0 = 0, i1 = 0, i2 = 0;
  for (int s = 0; s < S; ++s) {
    float dx = xyz2[((size_t)b * S + s) * 3 + 0] - px;
    float dy = xyz2[((size_t)b * S + s) * 3 + 1] - py;
    float dz = xyz2[((size_t)b * S + s) * 3 + 2] - pz;
    float d = dx * dx + dy * dy + dz * dz;
    if (d < d0)      { d2 = d1; i2 = i1; d1 = d0; i1 = i0; d0 = d; i0 = s; }
    else if (d < d1) { d2 = d1; i2 = i1; d1 = d;  i1 = s; }
    else if (d < d2) { d2 = d;  i2 = s; }
  }
  float w0 = 1.f / (d0 + 1e-8f), w1 = 1.f / (d1 + 1e-8f), w2 = 1.f / (d2 + 1e-8f);
  float ws = w0 + w1 + w2;
  w0 /= ws; w1 /= ws; w2 /= ws;
  const float* a = f2 + (size_t)i0 * C2;
  const float* bb = f2 + (size_t)i1 * C2;
  const float* cc = f2 + (size_t)i2 * C2;
  for (int c = 0; c < C2; ++c) out[c] = a[c] * w0 + bb[c] * w1 + cc[c] * w2;
}

// FP concat: X (B, C2+Cs, N) from interp (B,N,C2) and skip feat (B,N,Cs)
__global__ void concat_fp_kernel(const float* __restrict__ interp, const float* __restrict__ feat1,
                                 int N, int C2, int Cs, float* __restrict__ X) {
  int n = blockIdx.x * blockDim.x + threadIdx.x;
  int b = blockIdx.y;
  if (n >= N) return;
  int Cin = C2 + Cs;
  float* xb = X + (size_t)b * Cin * N;
  const float* ip = interp + ((size_t)b * N + n) * C2;
  for (int c = 0; c < C2; ++c) xb[(size_t)c * N + n] = ip[c];
  const float* fp = feat1 + ((size_t)b * N + n) * Cs;
  for (int c = 0; c < Cs; ++c) xb[(size_t)(C2 + c) * N + n] = fp[c];
}

// (B,C,N) -> (B,N,C)
__global__ void transpose_kernel(const float* __restrict__ Y, float* __restrict__ out,
                                 int C, int N) {
  int t = blockIdx.x * blockDim.x + threadIdx.x;
  int b = blockIdx.y;
  if (t >= C * N) return;
  int n = t / C, c = t % C;
  out[((size_t)b * N + n) * C + c] = Y[((size_t)b * C + c) * N + n];
}

__global__ void copy_kernel(const float* __restrict__ src, float* __restrict__ dst, int n) {
  int i = blockIdx.x * blockDim.x + threadIdx.x;
  if (i < n) dst[i] = src[i];
}

// ---------------------------------------------------------------------------
// Host orchestration
// ---------------------------------------------------------------------------
struct LayerP { const float *W, *b, *g, *be; };

static const float* run_mlp(const LayerP* ly, const int* ch, int nl,
                            const float* X, int L, float* y0, float* y1,
                            _Float16* Apack, _Float16* Bpack,
                            float* mean, float* var, hipStream_t stream) {
  const float* in = X;
  float* outs[2] = {y0, y1};
  for (int i = 0; i < nl; ++i) {
    int cin = ch[i], cout = ch[i + 1];
    int nks = (cin + 31) / 32;
    float* Y = outs[i & 1];

    int wtot = (cout / 16) * nks * 32;
    pack_w_kernel<<<(wtot + 255) / 256, 256, 0, stream>>>(ly[i].W, Apack, cin, cout, nks);

    int xtot = (L / 16) * nks * 32;
    dim3 gp((xtot + 255) / 256, B_SZ);
    pack_x_kernel<<<gp, 256, 0, stream>>>(in, Bpack, cin, L, nks);

    dim3 g((L / 32 + 3) / 4, cout / 32, B_SZ);
    gemm_wmma_kernel<<<g, 128, 0, stream>>>(Apack, Bpack, ly[i].b, Y, nks, cout, L);

    bn_stats_kernel<<<cout, 256, 0, stream>>>(Y, mean, var, cout, L, B_SZ);
    unsigned long long tot = (unsigned long long)B_SZ * cout * L;
    bn_relu_kernel<<<(unsigned)((tot + 255) / 256), 256, 0, stream>>>(
        Y, mean, var, ly[i].g, ly[i].be, cout, L, tot);
    in = Y;
  }
  return in;
}

extern "C" void kernel_launch(void* const* d_in, const int* in_sizes, int n_in,
                              void* d_out, int out_size, void* d_ws, size_t ws_size,
                              hipStream_t stream) {
  (void)in_sizes; (void)n_in; (void)out_size; (void)ws_size;

  // ---- input unpacking (setup_inputs dict insertion order) ----
  int cur = 0;
  const float* points = (const float*)d_in[cur++];
  LayerP sa[4][3], gl[2], fp[4][3];
  auto take = [&](LayerP& L) {
    L.W  = (const float*)d_in[cur++];
    L.b  = (const float*)d_in[cur++];
    L.g  = (const float*)d_in[cur++];
    L.be = (const float*)d_in[cur++];
  };
  for (int i = 0; i < 4; ++i) for (int j = 0; j < 3; ++j) take(sa[i][j]);
  for (int j = 0; j < 2; ++j) take(gl[j]);
  static const int fp_nl[4] = {2, 2, 3, 3};
  for (int i = 0; i < 4; ++i) for (int j = 0; j < fp_nl[i]; ++j) take(fp[i][j]);

  static const int sa_ch[4][4]  = {{4,32,32,64},{67,64,64,128},{131,128,128,256},{259,256,256,512}};
  static const int gl_ch[3]     = {515, 512, 1024};
  static const int fp_ch[4][4]  = {{1536,256,256,0},{512,256,128,0},{256,128,128,128},{129,128,128,128}};
  static const float radius[4]  = {0.5f, 1.f, 2.f, 4.f};
  static const int npts[5]      = {16384, 2048, 512, 128, 32};

  // ---- workspace bump allocator ----
  char* wp = (char*)d_ws;
  auto alloc = [&](size_t bytes) -> void* {
    void* r = (void*)wp;
    wp += (bytes + 255) & ~(size_t)255;
    return r;
  };
  float* xyzL[5];
  float* featL[5];
  int featC[5] = {1, 64, 128, 256, 512};
  for (int i = 0; i < 5; ++i) {
    xyzL[i]  = (float*)alloc((size_t)B_SZ * npts[i] * 3 * sizeof(float));
    featL[i] = (float*)alloc((size_t)B_SZ * npts[i] * featC[i] * sizeof(float));
  }
  int*      fpsidx  = (int*)alloc((size_t)B_SZ * 2048 * sizeof(int));
  int*      ballidx = (int*)alloc((size_t)B_SZ * 2048 * KNB * sizeof(int));
  float*    bufX    = (float*)alloc((size_t)4600000 * sizeof(float));
  float*    bufY0   = (float*)alloc((size_t)8500000 * sizeof(float));
  float*    bufY1   = (float*)alloc((size_t)8500000 * sizeof(float));
  float*    bufI    = (float*)alloc((size_t)4300000 * sizeof(float));
  float*    fpfeat  = (float*)alloc((size_t)1100000 * sizeof(float));
  _Float16* Apack   = (_Float16*)alloc((size_t)1100000 * sizeof(_Float16));
  _Float16* Bpack   = (_Float16*)alloc((size_t)6200000 * sizeof(_Float16));
  float*    mean    = (float*)alloc(1536 * sizeof(float));
  float*    var     = (float*)alloc(1536 * sizeof(float));
  float*    gfeat   = (float*)alloc((size_t)B_SZ * 1024 * sizeof(float));

  // ---- split points ----
  {
    dim3 g((N0 + 127) / 128, B_SZ);
    split_points_kernel<<<g, 128, 0, stream>>>(points, N0, xyzL[0], featL[0]);
  }

  // ---- set abstraction levels ----
  for (int i = 0; i < 4; ++i) {
    int N = npts[i], S = npts[i + 1];
    fps_kernel<<<B_SZ, FPS_T, 0, stream>>>(xyzL[i], N, S, fpsidx);
    {
      dim3 g((S + 127) / 128, B_SZ);
      gather_xyz_kernel<<<g, 128, 0, stream>>>(xyzL[i], fpsidx, N, S, xyzL[i + 1]);
      ballquery_kernel<<<g, 128, 0, stream>>>(xyzL[i], xyzL[i + 1], N, S, KNB,
                                              radius[i] * radius[i], ballidx);
    }
    int L = S * KNB;
    {
      dim3 g((L + 127) / 128, B_SZ);
      group_kernel<<<g, 128, 0, stream>>>(xyzL[i], featL[i], xyzL[i + 1], ballidx,
                                          N, S, KNB, featC[i], bufX);
    }
    const float* Y = run_mlp(sa[i], sa_ch[i], 3, bufX, L, bufY0, bufY1,
                             Apack, Bpack, mean, var, stream);
    int cout = sa_ch[i][3];
    dim3 g((S * cout + 127) / 128, B_SZ);
    maxpool_kernel<<<g, 128, 0, stream>>>(Y, featL[i + 1], cout, S, KNB);
  }

  // ---- global set abstraction ----
  {
    int N = npts[4];
    dim3 g((N + 127) / 128, B_SZ);
    concat_xyzfeat_kernel<<<g, 128, 0, stream>>>(xyzL[4], featL[4], N, featC[4], bufX);
    const float* Y = run_mlp(gl, gl_ch, 2, bufX, N, bufY0, bufY1,
                             Apack, Bpack, mean, var, stream);
    dim3 gc((1024 + 127) / 128, B_SZ);
    chanmax_kernel<<<gc, 128, 0, stream>>>(Y, gfeat, 1024, N);
  }

  // ---- feature propagation ----
  const float* src_feat = gfeat;
  const float* src_xyz  = xyzL[4];
  int srcC = 1024, srcS = 1;
  for (int i = 0; i < 4; ++i) {
    int tgt = 3 - i;
    int N = npts[tgt];
    dim3 g((N + 127) / 128, B_SZ);
    interp_kernel<<<g, 128, 0, stream>>>(xyzL[tgt], src_xyz, src_feat, N, srcS, srcC, bufI);
    concat_fp_kernel<<<g, 128, 0, stream>>>(bufI, featL[tgt], N, srcC, featC[tgt], bufX);
    const float* Y = run_mlp(fp[i], fp_ch[i], fp_nl[i], bufX, N, bufY0, bufY1,
                             Apack, Bpack, mean, var, stream);
    int outC = fp_ch[i][fp_nl[i]];
    dim3 gt(((size_t)N * outC + 127) / 128, B_SZ);
    if (i < 3) {
      transpose_kernel<<<gt, 128, 0, stream>>>(Y, fpfeat, outC, N);
      src_feat = fpfeat;
      src_xyz  = xyzL[tgt];
      srcC = outC;
      srcS = N;
    } else {
      transpose_kernel<<<gt, 128, 0, stream>>>(Y, (float*)d_out, outC, N);
    }
  }

  // ---- append global feature to output tail ----
  copy_kernel<<<(B_SZ * 1024 + 127) / 128, 128, 0, stream>>>(
      gfeat, (float*)d_out + (size_t)B_SZ * N0 * 128, B_SZ * 1024);
}
